// AxialAtten_17686675325050
// MI455X (gfx1250) — compile-verified, matrix operand order checked by reading, not run on hardware
//
#include <hip/hip_runtime.h>
#include <hip/hip_bf16.h>

// MI455X / gfx1250, wave32. Fused axial sigmoid-attention, f16 WMMA, f32 accum.

typedef __attribute__((ext_vector_type(16))) _Float16 v16h;
typedef __attribute__((ext_vector_type(8)))  _Float16 v8h;
typedef __attribute__((ext_vector_type(8)))  float    v8f;

union F16Frag { v16h v; v8h h[2]; };

#define NB 4
#define NN 4096   // sequence length per batch (W*C or H*C)
#define DD 64     // feature dim (H or W)
#define ELEMS_PER_BATCH 262144  // 64*64*64
#define MCHUNKS (NN / 256)      // m-iterations per wave (8 waves x 32 each)

__device__ __forceinline__ v8f wmma16(const F16Frag& a, const F16Frag& b, v8f c) {
  return __builtin_amdgcn_wmma_f32_16x16x32_f16(false, a.v, false, b.v,
                                                (short)0, c, false, false);
}

// sigmoid(s * 0.125) with minimal VALU ops
__device__ __forceinline__ float fast_sigmoid8(float s) {
#if __has_builtin(__builtin_amdgcn_tanhf)
  // sigmoid(x) = 0.5*tanh(x/2) + 0.5 ; x = s/8  ->  tanh(s/16)
  return __builtin_fmaf(0.5f, __builtin_amdgcn_tanhf(s * 0.0625f), 0.5f);
#else
  // exp(-s/8) = exp2(s * (-0.125*log2 e)) ; single fused scale constant
  const float c = -0.125f * 1.44269504088896340736f;
  return __builtin_amdgcn_rcpf(1.0f + __builtin_amdgcn_exp2f(s * c));
#endif
}

// ---------------------------------------------------------------------------
// Projection: Q[n,o] = X[n,:]·Wq[o,:] + bq ; Vt[o,n] = X[n,:]·Wv[o,:] + bv
// mode 0 (height axis): X[n,t] = src[b*262144 + t*4096 + n]
// mode 1 (width  axis): X[n,t] = src[b*262144 + (n>>6)*4096 + t*64 + (n&63)]
// ---------------------------------------------------------------------------
__global__ __launch_bounds__(256) void proj_kernel(
    const float* __restrict__ src,
    const float* __restrict__ wq, const float* __restrict__ bq,
    const float* __restrict__ wv, const float* __restrict__ bv,
    _Float16* __restrict__ Q, _Float16* __restrict__ Vt, int mode)
{
  __shared__ float sX[4][64];
  __shared__ float sWq[64][65];   // transposed, padded vs bank conflicts
  __shared__ float sWv[64][65];

  const int tid = threadIdx.x;
  const int b   = blockIdx.y;
  const int n0  = blockIdx.x * 4;

  for (int i = tid; i < 4096; i += 256) {
    int o = i >> 6, t = i & 63;
    sWq[t][o] = wq[i];
    sWv[t][o] = wv[i];
  }
  {
    int r = tid >> 6, t = tid & 63;
    int n = n0 + r;
    const float* xb = src + (size_t)b * ELEMS_PER_BATCH;
    sX[r][t] = (mode == 0) ? xb[t * 4096 + n]
                           : xb[(n >> 6) * 4096 + t * 64 + (n & 63)];
  }
  __syncthreads();

  const int o = tid & 63;
  const int r = tid >> 6;
  const int n = n0 + r;
  float aq = bq[o], av = bv[o];
#pragma unroll 8
  for (int k = 0; k < 64; ++k) {
    float xv = sX[r][k];
    aq += xv * sWq[k][o];
    av += xv * sWv[k][o];
  }
  Q [((size_t)b * NN + n) * DD + o] = (_Float16)aq;
  Vt[((size_t)b * DD + o) * NN + n] = (_Float16)av;
}

// ---------------------------------------------------------------------------
// Fused streaming sigmoid-attention:
//   O[n,:] = sum_m sigmoid(Q[n]·Q[m]/8) * V[m,:]
//   dst = O * weight + res   (scatter per `mode`)
// Block: 256 thr = 8 waves; owns 16 rows x 64 features. Waves stride over m.
// ---------------------------------------------------------------------------
__global__ __launch_bounds__(256) void atten_kernel(
    const _Float16* __restrict__ Q, const _Float16* __restrict__ Vt,
    const float* __restrict__ res, float* __restrict__ dst,
    const float* __restrict__ weight, int mode)
{
  __shared__ __align__(16) _Float16 sP[8][16][32];  // per-wave P staging (D->A layout)
  __shared__ float sO[8][16][64];                   // per-wave partial O

  const int tid  = threadIdx.x;
  const int wave = tid >> 5;
  const int lane = tid & 31;
  const int hi   = lane >> 4;   // lane group selects K interleave half
  const int lo   = lane & 15;   // row (A/B frag) or column (C/D frag)
  const int b    = blockIdx.y;
  const int n0   = blockIdx.x * 16;

  const _Float16* Qb = Q  + (size_t)b * NN * DD;
  const _Float16* Vb = Vt + (size_t)b * DD * NN;

  // A fragments of the Q_n tile (16 rows x K=64), resident for whole kernel.
  // A-frag layout: lane holds row lo; 16 halves cover K = {8*hi+0..7, 16+8*hi+0..7} (+chunk).
  const _Float16* qrow = Qb + (size_t)(n0 + lo) * DD;
  F16Frag A0, A1;
  A0.h[0] = *(const v8h*)(qrow + 8 * hi);
  A0.h[1] = *(const v8h*)(qrow + 16 + 8 * hi);
  A1.h[0] = *(const v8h*)(qrow + 32 + 8 * hi);
  A1.h[1] = *(const v8h*)(qrow + 48 + 8 * hi);

  v8f O[4];
#pragma unroll
  for (int t = 0; t < 4; ++t) O[t] = (v8f){};

  // uniform literal trip count -> scalar loop control (no exec-mask loop)
  for (int it = 0; it < MCHUNKS; ++it) {
    const int m0 = it * 256 + wave * 32;

    // ---- issue ALL global loads for this chunk first (one clause,      ----
    // ---- lets V loads overlap the S-WMMAs and the sigmoid VALU tail)   ----
    F16Frag B[4];   // Q_m^T B-frags: [j*2+kc], j = N-tile, kc = K-chunk
    const _Float16* brow0 = Qb + (size_t)(m0 + lo) * DD;
    const _Float16* brow1 = Qb + (size_t)(m0 + 16 + lo) * DD;
    B[0].h[0] = *(const v8h*)(brow0 + 8 * hi);
    B[0].h[1] = *(const v8h*)(brow0 + 16 + 8 * hi);
    B[1].h[0] = *(const v8h*)(brow0 + 32 + 8 * hi);
    B[1].h[1] = *(const v8h*)(brow0 + 48 + 8 * hi);
    B[2].h[0] = *(const v8h*)(brow1 + 8 * hi);
    B[2].h[1] = *(const v8h*)(brow1 + 16 + 8 * hi);
    B[3].h[0] = *(const v8h*)(brow1 + 32 + 8 * hi);
    B[3].h[1] = *(const v8h*)(brow1 + 48 + 8 * hi);

    F16Frag BV[4];  // V B-frags from V^T rows, 4 feature tiles
#pragma unroll
    for (int t = 0; t < 4; ++t) {
      const _Float16* vrow = Vb + (size_t)(t * 16 + lo) * NN + m0;
      BV[t].h[0] = *(const v8h*)(vrow + 8 * hi);
      BV[t].h[1] = *(const v8h*)(vrow + 16 + 8 * hi);
    }

    // ---- S = Q_n (16x64) @ Q_m^T (64x32), two 16x16 N-tiles, K chained ----
    v8f S0 = (v8f){}, S1 = (v8f){};
    S0 = wmma16(A0, B[0], S0);
    S0 = wmma16(A1, B[1], S0);
    S1 = wmma16(A0, B[2], S1);
    S1 = wmma16(A1, B[3], S1);

    // sigmoid(S/8) -> f16, staged to LDS (C/D layout: row = r+8*hi, col = lo)
#pragma unroll
    for (int rr = 0; rr < 8; ++rr) {
      sP[wave][rr + 8 * hi][lo]      = (_Float16)fast_sigmoid8(S0[rr]);
      sP[wave][rr + 8 * hi][16 + lo] = (_Float16)fast_sigmoid8(S1[rr]);
    }
    asm volatile("s_wait_dscnt 0" ::: "memory");

    // Re-read P as an A fragment (16 rows x K=32)
    F16Frag PA;
    const _Float16* prow = &sP[wave][lo][0];
    PA.h[0] = *(const v8h*)(prow + 8 * hi);
    PA.h[1] = *(const v8h*)(prow + 16 + 8 * hi);

    // ---- O += P (16x32) @ V (32x64), V frags already resident ----
#pragma unroll
    for (int t = 0; t < 4; ++t)
      O[t] = wmma16(PA, BV[t], O[t]);

    // prefetch next chunk of Q (uniform guard)
    if (it + 1 < MCHUNKS)
      __builtin_prefetch(Qb + (size_t)(m0 + 256 + lo) * DD, 0, 0);
  }

  // ---- cross-wave reduction of O, residual + weight, scatter store ----
#pragma unroll
  for (int t = 0; t < 4; ++t)
#pragma unroll
    for (int rr = 0; rr < 8; ++rr)
      sO[wave][rr + 8 * hi][t * 16 + lo] = O[t][rr];
  __syncthreads();

  const float wgt = weight[0];
  for (int i = tid; i < 1024; i += 256) {
    int row = i >> 6, d = i & 63;
    float s = 0.0f;
#pragma unroll
    for (int w = 0; w < 8; ++w) s += sO[w][row][d];
    int n = n0 + row;
    size_t idx = (mode == 0)
        ? ((size_t)b * ELEMS_PER_BATCH + (size_t)d * 4096 + n)
        : ((size_t)b * ELEMS_PER_BATCH + (size_t)(n >> 6) * 4096 + d * 64 + (n & 63));
    dst[idx] = s * wgt + res[idx];
  }
}

// ---------------------------------------------------------------------------
extern "C" void kernel_launch(void* const* d_in, const int* in_sizes, int n_in,
                              void* d_out, int out_size, void* d_ws, size_t ws_size,
                              hipStream_t stream) {
  const float* x        = (const float*)d_in[0];
  const float* hq_w     = (const float*)d_in[1];
  const float* hq_b     = (const float*)d_in[2];
  const float* hv_w     = (const float*)d_in[3];
  const float* hv_b     = (const float*)d_in[4];
  const float* wq_w     = (const float*)d_in[5];
  const float* wq_b     = (const float*)d_in[6];
  const float* wv_w     = (const float*)d_in[7];
  const float* wv_b     = (const float*)d_in[8];
  const float* h_weight = (const float*)d_in[9];
  const float* w_weight = (const float*)d_in[10];

  // workspace: x2 (4 MB f32) | Q (2 MB f16) | Vt (2 MB f16)
  float*     x2 = (float*)d_ws;
  _Float16*  Q  = (_Float16*)((char*)d_ws + (4u << 20));
  _Float16*  Vt = (_Float16*)((char*)d_ws + (6u << 20));
  float*     out = (float*)d_out;

  dim3 pb(256), pg(NN / 4, NB);
  dim3 ab(256), ag(NN / 16, NB);

  // height axis: feature dim = H
  proj_kernel <<<pg, pb, 0, stream>>>(x,  hq_w, hq_b, hv_w, hv_b, Q, Vt, 0);
  atten_kernel<<<ag, ab, 0, stream>>>(Q, Vt, x,  x2,  h_weight, 0);
  // width axis: feature dim = W
  proj_kernel <<<pg, pb, 0, stream>>>(x2, wq_w, wq_b, wv_w, wv_b, Q, Vt, 1);
  atten_kernel<<<ag, ab, 0, stream>>>(Q, Vt, x2, out, w_weight, 1);
}
